// Residual_block_33337536152141
// MI455X (gfx1250) — matrix operand order repaired and checked by reading, not compiled, not run on hardware
//
#include <hip/hip_runtime.h>
#include <hip/hip_bf16.h>
#include <math.h>

// ---------------------------------------------------------------------------
// CDNA5 (gfx1250) implementation of the SSSD residual block.
// Heavy ops run on v_wmma_f32_16x16x32_bf16 (wave32, f32 accumulate).
// All WMMA operands are built from zero-padded, fragment-ordered LDS tiles so
// the inner loops contain only ds_load_b128 / ds_load_u16 + v_wmma (no
// per-element predication, no EXEC juggling).
// ---------------------------------------------------------------------------

typedef __attribute__((ext_vector_type(16))) __bf16 v16bf;
typedef __attribute__((ext_vector_type(8)))  float  v8f;

static __device__ __forceinline__ __bf16 f2bf(float f) {
  union { float f; unsigned u; } c; c.f = f;
  unsigned r = c.u + 0x7FFFu + ((c.u >> 16) & 1u);     // round-to-nearest-even
  unsigned short h = (unsigned short)(r >> 16);
  union { unsigned short s; __bf16 b; } o; o.s = h;
  return o.b;
}

// A-operand per-lane K order for 16-bit 16x32 WMMA fragments: contiguous k = 16*h + e (matches B)
static __device__ __forceinline__ int a_kmap(int e, int h) {
  return 16 * h + e;
}

// ---------------------------------------------------------------------------
// FiLM: part[b,c] = demb[b,:] @ Wfc[:,c] + bfc[c];  h0 = x + part (broadcast L)
// grid = (C, B), block = 256
// ---------------------------------------------------------------------------
__global__ __launch_bounds__(256) void film_kernel(
    const float* __restrict__ x, const float* __restrict__ demb,
    const float* __restrict__ Wfc, const float* __restrict__ bfc,
    float* __restrict__ h0, int C, int L, int E) {
  const int c = blockIdx.x, b = blockIdx.y, tid = threadIdx.x;
  __shared__ float red[256];
  float s = 0.f;
  for (int e = tid; e < E; e += 256) s += demb[b * E + e] * Wfc[(size_t)e * C + c];
  red[tid] = s;
  __syncthreads();
  for (int off = 128; off > 0; off >>= 1) {
    if (tid < off) red[tid] += red[tid + off];
    __syncthreads();
  }
  const float part = red[0] + bfc[c];
  const size_t base = ((size_t)b * C + c) * L;
  for (int l = tid; l < L; l += 256) h0[base + l] = x[base + l] + part;
}

// ---------------------------------------------------------------------------
// Generic causal k-tap conv / 1x1 projection as bf16 WMMA GEMM.
//   Y[b,o,l] = sum_{t,ci} W[t,ci,o] * X[b,ci,l-(taps-1)+t] + bias[o]
// Block = 128 threads (4 waves), 32x32 output tile (2x2 wave sub-tiles).
// K loop over 32-channel chunks; weights staged in A-fragment order, X staged
// with a 2-position causal halo. Fragment fetch = one 32B LDS vector load.
// grid = (L/32, Cout/32, B), block = 128.
// ---------------------------------------------------------------------------
#define GM 32
#define GN 32
#define HALO 2

__global__ __launch_bounds__(128) void wmma_conv_gemm(
    const float* __restrict__ X, const float* __restrict__ W,
    const float* __restrict__ bias, float* __restrict__ Y,
    int L, int Cin, int Cout, int taps) {
  __shared__ alignas(32) __bf16 Ap[3][GM][32];       // [tap][outch][frag pos]
  __shared__ alignas(32) __bf16 Xp[GN + HALO][32];   // [pos][k], rows 64B

  const int tid = threadIdx.x;
  const int lane = tid & 31, wave = tid >> 5;
  const int l0 = blockIdx.x * GN, o0 = blockIdx.y * GM, b = blockIdx.z;
  const int hl = lane >> 4, mrow = lane & 15, ncol = lane & 15;
  const int mt = wave >> 1, nt = wave & 1;           // 2x2 wave tiling

  // X staging assignment: 4 threads per channel row, stride over positions
  const int krow = tid >> 2;        // 0..31
  const int j0 = tid & 3;

  v8f acc = {};
  for (int kc = 0; kc * 32 < Cin; ++kc) {
    __syncthreads();   // protect previous iteration's fragment reads

    // ---- stage weights in fragment order: Ap[t][o][16*h + e] = W[t, k(e,h), o]
    for (int id = tid; id < taps * GM * 32; id += 128) {
      const int o = id & (GM - 1);
      const int s = (id >> 5) & 31;
      const int t = id >> 10;
      const int k = a_kmap(s & 15, s >> 4);
      const int ci = kc * 32 + k;
      const int cic = ci < Cin ? ci : Cin - 1;                    // clamp
      const float wv = W[((size_t)t * Cin + cic) * Cout + (o0 + o)];
      Ap[t][o][s] = f2bf(ci < Cin ? wv : 0.f);                    // select
    }

    // ---- stage activations with causal halo: Xp[j][k] = X[b, ci, l0-HALO+j]
    {
      const int ci = kc * 32 + krow;
      const int cic = ci < Cin ? ci : Cin - 1;                    // clamp
      const size_t rowb = ((size_t)b * Cin + cic) * L;
      if ((kc + 1) * 32 < Cin)
        __builtin_prefetch(&X[rowb + 32 * (size_t)L + l0], 0, 1); // next chunk
      for (int j = j0; j < GN + HALO; j += 4) {
        const int l = l0 - HALO + j;
        const int lc = (l < 0) ? 0 : ((l >= L) ? L - 1 : l);      // clamp
        const float xv = X[rowb + lc];
        Xp[j][krow] = f2bf((ci < Cin && l >= 0 && l < L) ? xv : 0.f);
      }
    }
    __syncthreads();

    // ---- fragments: one aligned 32B LDS vector load each, then WMMA
    for (int t = 0; t < taps; ++t) {
      const v16bf a = *(const v16bf*)&Ap[t][16 * mt + mrow][16 * hl];
      const int j = 16 * nt + ncol + t + (HALO - (taps - 1));
      const v16bf bm = *(const v16bf*)&Xp[j][16 * hl];
      acc = __builtin_amdgcn_wmma_f32_16x16x32_bf16(
          false, a, false, bm, (short)0, acc, false, false);
    }
  }

  // C/D layout: VGPR r -> M = r + 8*half, N = lane&15
  #pragma unroll
  for (int r = 0; r < 8; ++r) {
    const int m = 16 * mt + r + 8 * hl;
    const int n = 16 * nt + ncol;
    Y[((size_t)b * Cout + o0 + m) * L + l0 + n] = acc[r] + bias[o0 + m];
  }
}

// ---------------------------------------------------------------------------
// S4 long causal conv:  y[b,f,l] = sum_{tau<=l} K[f,tau] * x[b,f,l-tau]
// Toeplitz-block WMMA: y_tile[i] = sum_d T_d(f) x_tile[i-d],
//   T_d[m,c] = K[f, 16d + m - c].  K-dim 32 covers two adjacent input tiles.
// Block = 1 channel f; 8 waves; x[8 batches] and K[f] staged to LDS as bf16
// with 16-element zero front-pads, so the inner loop is fully unguarded:
// B operand = one aligned 32B ds vector load, A = 16 ds_load_u16, then WMMA.
// grid = (F), block = 256.
// ---------------------------------------------------------------------------
#define LSEQ 2048
#define PAD 16

__global__ __launch_bounds__(256) void s4_toeplitz_wmma(
    const float* __restrict__ Xin, const float* __restrict__ Kker,
    float* __restrict__ Yconv, int F) {
  __shared__ alignas(32) __bf16 xs[8][PAD + LSEQ];   // rows 4128B (32B mult.)
  __shared__ alignas(32) __bf16 ks[PAD + LSEQ];
  const int f = blockIdx.x;
  const int tid = threadIdx.x;

  for (int bb = 0; bb < 8; ++bb) {
    const size_t base = ((size_t)bb * F + f) * LSEQ;
    for (int i = tid; i < PAD + LSEQ; i += 256) {
      const int l = i - PAD;
      const float xv = Xin[base + (l < 0 ? 0 : l)];                // clamp
      xs[bb][i] = f2bf(l >= 0 ? xv : 0.f);                         // select
    }
  }
  {
    const size_t base = (size_t)f * LSEQ;
    for (int i = tid; i < PAD + LSEQ; i += 256) {
      const int l = i - PAD;
      const float kv = Kker[base + (l < 0 ? 0 : l)];
      ks[i] = f2bf(l >= 0 ? kv : 0.f);
    }
  }
  __syncthreads();

  const int lane = tid & 31, wave = tid >> 5;
  const int hl = lane >> 4;
  const int mrow = lane & 15, ncol = lane & 15;
  const int bcol = ncol & 7, itadd = ncol >> 3;

  for (int g = wave; g < (LSEQ / 32); g += 8) {     // 64 column groups
    v8f acc = {};
    const int icol = 2 * g + itadd;                 // output tile per column
    for (int p = 0; p <= g; ++p) {                  // uniform trip count
      const int shift = 16 * (2 * p + 1);
      v16bf a;
      #pragma unroll
      for (int e = 0; e < 16; ++e) {                // Toeplitz A from K[f]
        a[e] = ks[PAD + shift + mrow - a_kmap(e, hl)];  // pad covers idx>=-15
      }
      const int s0 = icol - 2 * p - 1;              // base input tile (>= -1)
      const v16bf bm = *(const v16bf*)&xs[bcol][PAD + 16 * (s0 + hl)];
      acc = __builtin_amdgcn_wmma_f32_16x16x32_bf16(
          false, a, false, bm, (short)0, acc, false, false);
    }
    #pragma unroll
    for (int r = 0; r < 8; ++r) {
      const int m = r + 8 * hl;
      Yconv[((size_t)bcol * F + f) * LSEQ + 16 * icol + m] = acc[r];
    }
  }
}

// ---------------------------------------------------------------------------
// y = conv + (D[f]+1)*x, then LayerNorm over F channels, scale/shift.
// One block per (b,l) column. block = 64, each thread covers 8 channels.
// ---------------------------------------------------------------------------
__global__ __launch_bounds__(64) void s4_ln_kernel(
    const float* __restrict__ Yconv, const float* __restrict__ Xres,
    const float* __restrict__ D, const float* __restrict__ gam,
    const float* __restrict__ bet, float* __restrict__ Out, int F, int L) {
  const int col = blockIdx.x;
  const int b = col / L, l = col - b * L;
  const int tid = threadIdx.x;
  __shared__ float ssum[64], ssq[64];
  float vals[8];
  float s = 0.f, sq = 0.f;
  #pragma unroll
  for (int j = 0; j < 8; ++j) {
    const int f = tid + 64 * j;
    const size_t idx = ((size_t)b * F + f) * L + l;
    const float y = Yconv[idx] + (D[f] + 1.0f) * Xres[idx];
    vals[j] = y; s += y; sq += y * y;
  }
  ssum[tid] = s; ssq[tid] = sq;
  __syncthreads();
  for (int off = 32; off > 0; off >>= 1) {
    if (tid < off) { ssum[tid] += ssum[tid + off]; ssq[tid] += ssq[tid + off]; }
    __syncthreads();
  }
  const float mu = ssum[0] / F;
  const float var = ssq[0] / F - mu * mu;
  const float rs = rsqrtf(var + 1e-5f);
  #pragma unroll
  for (int j = 0; j < 8; ++j) {
    const int f = tid + 64 * j;
    const size_t idx = ((size_t)b * F + f) * L + l;
    Out[idx] = gam[f] * (vals[j] - mu) * rs + bet[f];
  }
}

// ---------------------------------------------------------------------------
// Elementwise helpers
// ---------------------------------------------------------------------------
__global__ void add_inplace_kernel(float* __restrict__ a,
                                   const float* __restrict__ bsrc, size_t n) {
  size_t i = (size_t)blockIdx.x * blockDim.x + threadIdx.x;
  if (i < n) a[i] += bsrc[i];
}

__global__ void gate_kernel(const float* __restrict__ h, float* __restrict__ out,
                            int C, int L, size_t n) {
  size_t i = (size_t)blockIdx.x * blockDim.x + threadIdx.x;
  if (i >= n) return;
  const int l = (int)(i % L);
  const size_t t2 = i / L;
  const int c = (int)(t2 % C);
  const int b = (int)(t2 / C);
  const size_t base = ((size_t)b * (2 * C)) * L;
  const float th = tanhf(h[base + (size_t)c * L + l]);
  const float sg = h[base + (size_t)(c + C) * L + l];
  out[i] = th * (1.0f / (1.0f + expf(-sg)));
}

__global__ void combine_kernel(const float* __restrict__ x,
                               const float* __restrict__ res,
                               float* __restrict__ out, size_t n) {
  size_t i = (size_t)blockIdx.x * blockDim.x + threadIdx.x;
  if (i < n) out[i] = (x[i] + res[i]) * 0.70710678118654752f;
}

// ---------------------------------------------------------------------------
// Driver. Input order = setup_inputs() dict order (all float32).
// ---------------------------------------------------------------------------
extern "C" void kernel_launch(void* const* d_in, const int* in_sizes, int n_in,
                              void* d_out, int out_size, void* d_ws, size_t ws_size,
                              hipStream_t stream) {
  const float* x      = (const float*)d_in[0];
  const float* cond   = (const float*)d_in[1];
  const float* demb   = (const float*)d_in[2];
  const float* W_fc   = (const float*)d_in[3];
  const float* b_fc   = (const float*)d_in[4];
  const float* conv_w = (const float*)d_in[5];
  const float* conv_b = (const float*)d_in[6];
  const float* K1     = (const float*)d_in[7];
  const float* D1     = (const float*)d_in[8];
  const float* g1     = (const float*)d_in[9];
  const float* beta1  = (const float*)d_in[10];
  const float* cond_w = (const float*)d_in[11];
  const float* cond_b = (const float*)d_in[12];
  const float* K2     = (const float*)d_in[13];
  const float* D2     = (const float*)d_in[14];
  const float* g2     = (const float*)d_in[15];
  const float* beta2  = (const float*)d_in[16];
  const float* res_w  = (const float*)d_in[17];
  const float* res_b  = (const float*)d_in[18];
  const float* skip_w = (const float*)d_in[19];
  const float* skip_b = (const float*)d_in[20];
  (void)in_sizes; (void)n_in; (void)out_size; (void)ws_size;

  const int B = 8, C = 256, C2 = 512, L = 2048, E = 512, CIN2 = 28, SK = 256;
  const size_t BCL  = (size_t)B * C * L;    // 4,194,304
  const size_t BC2L = (size_t)B * C2 * L;   // 8,388,608

  float* buf0 = (float*)d_ws;               // (B,C,L)  16 MB
  float* buf1 = buf0 + BCL;                 // (B,C2,L) 32 MB
  float* buf2 = buf1 + BC2L;                // (B,C2,L) 32 MB
  float* outp = (float*)d_out;

  // 1) FiLM bias + add -> buf0
  film_kernel<<<dim3(C, B), 256, 0, stream>>>(x, demb, W_fc, b_fc, buf0, C, L, E);

  // 2) causal conv k=3 (256 -> 512) -> buf1   [WMMA]
  wmma_conv_gemm<<<dim3(L / GN, C2 / GM, B), 128, 0, stream>>>(
      buf0, conv_w, conv_b, buf1, L, C, C2, 3);

  // 3) S4 layer 1: long conv (Toeplitz WMMA) -> buf2, then LN (in-place buf2)
  s4_toeplitz_wmma<<<dim3(C2), 256, 0, stream>>>(buf1, K1, buf2, C2);
  s4_ln_kernel<<<dim3(B * L), 64, 0, stream>>>(buf2, buf1, D1, g1, beta1, buf2, C2, L);

  // 4) conditioning 1x1 (28 -> 512) -> buf1, add into buf2   [WMMA]
  wmma_conv_gemm<<<dim3(L / GN, C2 / GM, B), 128, 0, stream>>>(
      cond, cond_w, cond_b, buf1, L, CIN2, C2, 1);
  add_inplace_kernel<<<dim3((unsigned)((BC2L + 255) / 256)), 256, 0, stream>>>(
      buf2, buf1, BC2L);

  // 5) S4 layer 2 -> buf1 (conv), LN (in-place buf1)
  s4_toeplitz_wmma<<<dim3(C2), 256, 0, stream>>>(buf2, K2, buf1, C2);
  s4_ln_kernel<<<dim3(B * L), 64, 0, stream>>>(buf1, buf2, D2, g2, beta2, buf1, C2, L);

  // 6) gate: tanh(h[:C]) * sigmoid(h[C:]) -> buf0
  gate_kernel<<<dim3((unsigned)((BCL + 255) / 256)), 256, 0, stream>>>(
      buf1, buf0, C, L, BCL);

  // 7) res 1x1 (256->256) -> buf2, combine with x -> d_out[0 : BCL]   [WMMA]
  wmma_conv_gemm<<<dim3(L / GN, C / GM, B), 128, 0, stream>>>(
      buf0, res_w, res_b, buf2, L, C, C, 1);
  combine_kernel<<<dim3((unsigned)((BCL + 255) / 256)), 256, 0, stream>>>(
      x, buf2, outp, BCL);

  // 8) skip 1x1 (256->256) -> d_out[BCL : 2*BCL]   [WMMA]
  wmma_conv_gemm<<<dim3(L / GN, SK / GM, B), 128, 0, stream>>>(
      buf0, skip_w, skip_b, outp + BCL, L, C, SK, 1);
}
